// GraphEncoder_71107478553359
// MI455X (gfx1250) — matrix operand order repaired and checked by reading, not compiled, not run on hardware
//
#include <hip/hip_runtime.h>
#include <hip/hip_bf16.h>
#include <math.h>

// ---------------------------------------------------------------------------
// GraphEncoder forward for MI455X (gfx1250, wave32).
//   D=256, B=4, N=2048, IN=16, POS=3, LAT=128, DEPTH=3, NBLK=8
// Dense GEMMs: v_wmma_f32_16x16x32_f16 with f32 accumulate.
//   - W pre-packed (per launch) into per-lane B-fragment order (f16): each
//     lane loads its whole B fragment as 32 contiguous bytes.
//   - A tile staged into LDS already converted to f16 (single conversion per
//     block, padded rows -> conflict-light ds_load_b128 fragment reads).
// Binary-adjacency einsum done as exact sparse neighbor gather-sum, with the
// gathered adjacency slab copied global->LDS via the async path (ASYNCcnt).
// ---------------------------------------------------------------------------

typedef __attribute__((ext_vector_type(16))) _Float16 v16h;
typedef __attribute__((ext_vector_type(8)))  _Float16 v8h;
typedef __attribute__((ext_vector_type(4)))  _Float16 v4h;
typedef __attribute__((ext_vector_type(8)))  float    v8f;

#define GE_B    4
#define GE_N    2048
#define GE_D    256
#define GE_IN   16
#define GE_LAT  128
#define GE_LDAH 264      // padded f16 LDS row stride: 528B = 16B mod 256B
#define GE_WPK  65536    // packed weight elements (8*256*2*16 f16)

#if __has_builtin(__builtin_amdgcn_global_load_async_to_lds_b32) && \
    __has_builtin(__builtin_amdgcn_s_wait_asynccnt)
#define GE_ASYNC_B32 1
#else
#define GE_ASYNC_B32 0
#endif

// ---------------------------------------------------------------------------
__global__ __launch_bounds__(256) void k_init_cidx(int* __restrict__ c) {
    int idx = blockIdx.x * 256 + threadIdx.x;   // B*N threads
    c[idx] = idx & (GE_N - 1);
}

// ---------------------------------------------------------------------------
// Positional stats: per-column mean and 1/(std+1e-8) of tiled pos.
__global__ __launch_bounds__(192) void k_pos_stats(const float* __restrict__ pos,
                                                   float* __restrict__ st) {
    __shared__ float red[3][64];
    int t = threadIdx.x, c = t >> 6, l = t & 63;
    float s = 0.f;
    for (int i = l; i < GE_N; i += 64) s += pos[i * 3 + c];
    red[c][l] = s; __syncthreads();
    for (int w = 32; w > 0; w >>= 1) { if (l < w) red[c][l] += red[c][l + w]; __syncthreads(); }
    float mean = red[c][0] * (1.f / GE_N);
    __syncthreads();
    float ss = 0.f;
    for (int i = l; i < GE_N; i += 64) { float d = pos[i * 3 + c] - mean; ss += d * d; }
    red[c][l] = ss; __syncthreads();
    for (int w = 32; w > 0; w >>= 1) { if (l < w) red[c][l] += red[c][l + w]; __syncthreads(); }
    if (l == 0) {
        float SS = red[c][0] * (float)GE_B;                    // tiled B times
        float sd = sqrtf(SS / ((float)GE_B * GE_N - 1.f));     // ddof=1
        st[c] = mean; st[3 + c] = 1.f / (sd + 1e-8f);
    }
}

// pos MLP layer 1 (K=3) + SiLU -> u (N, 256)
__global__ __launch_bounds__(256) void k_pos_mlp1(const float* __restrict__ pos,
                                                  const float* __restrict__ st,
                                                  const float* __restrict__ pw1,
                                                  const float* __restrict__ pb1,
                                                  float* __restrict__ u) {
    int i = blockIdx.x, d = threadIdx.x;
    float acc = pb1[d];
#pragma unroll
    for (int c = 0; c < 3; ++c) {
        float pn = (pos[i * 3 + c] - st[c]) * st[3 + c];
        acc += pn * pw1[c * GE_D + d];
    }
    u[(size_t)i * GE_D + d] = acc * (1.f / (1.f + __expf(-acc)));   // SiLU
}

// ---------------------------------------------------------------------------
// Pack W (256x256 f32 row-major) into per-lane B-fragment order, f16:
//   Wp[ks*8192 + n*32 + g*16 + j] = W[(ks*32 + g*16 + j)*256 + n]
__global__ __launch_bounds__(256) void k_pack_w(const float* __restrict__ W,
                                                _Float16* __restrict__ Wp) {
    int idx = blockIdx.x * 256 + threadIdx.x;   // 65536 elements
    int j  = idx & 15;
    int g  = (idx >> 4) & 1;
    int n  = (idx >> 5) & 255;
    int ks = idx >> 13;
    Wp[idx] = (_Float16)W[(size_t)(ks * 32 + g * 16 + j) * GE_D + n];
}

// ---------------------------------------------------------------------------
// WMMA GEMM: C[M,256] = act(A[M,256] @ W[256,256] + bias) (+ resid)
// Wp: packed f16 weights (k_pack_w). act: 0=none, 1=SiLU.
// Block = 128 threads (4 waves), wave -> one 16x16 C tile. grid = (M/16, 4).
__global__ __launch_bounds__(128)
void k_gemm256(const float* __restrict__ A, const _Float16* __restrict__ Wp,
               const float* __restrict__ bias, const float* __restrict__ resid,
               float* __restrict__ C, int M, int act) {
    __shared__ _Float16 Ash[16 * GE_LDAH];      // padded f16 A tile (~8.3 KB)
    const int tid  = threadIdx.x;
    const int row0 = blockIdx.x * 16;

    // Stage A tile with a single f32->f16 conversion per element.
    for (int ch = tid; ch < 16 * 64; ch += 128) {
        int r = ch >> 6, c4 = (ch & 63) << 2;
        float4 f = *(const float4*)(A + (size_t)(row0 + r) * GE_D + c4);
        v4h hv = { (_Float16)f.x, (_Float16)f.y, (_Float16)f.z, (_Float16)f.w };
        *(v4h*)(&Ash[r * GE_LDAH + c4]) = hv;   // ds_store_b64
    }
    __syncthreads();

    const int wave = tid >> 5;
    const int lane = tid & 31;                  // wave32
    const int m16  = lane & 15;
    const int grp  = lane >> 4;
    const int n0   = (blockIdx.y * 4 + wave) * 16;
    const int ncol = n0 + m16;
    const _Float16* arow = &Ash[m16 * GE_LDAH];

    v8f acc = {};
#pragma unroll
    for (int ks = 0; ks < 8; ++ks) {
        const int kb = ks * 32;
        // A frag (contiguous k, matches packed B): lanes 0-15 K=0..15; 16-31 K=16..31
        // = two contiguous 16-byte LDS reads, concatenated.
        v8h lo = *(const v8h*)(arow + kb + grp * 16);
        v8h hi = *(const v8h*)(arow + kb + grp * 16 + 8);
        v16h a = __builtin_shufflevector(lo, hi, 0, 1, 2, 3, 4, 5, 6, 7,
                                                 8, 9, 10, 11, 12, 13, 14, 15);
        // B frag: one 32-byte contiguous vector per lane from packed weights
        const v16h b = *(const v16h*)(Wp + (size_t)ks * 8192 + ncol * 32 + grp * 16);
        if (ks < 7)  // gfx1250 global_prefetch_b8 of next packed slab
            __builtin_prefetch(Wp + (size_t)(ks + 1) * 8192 + ncol * 32 + grp * 16, 0, 3);
        acc = __builtin_amdgcn_wmma_f32_16x16x32_f16(
            false, a, false, b, (short)0, acc, false, false);
    }

    float bs = bias[ncol];
#pragma unroll
    for (int r = 0; r < 8; ++r) {
        int mr = row0 + r + grp * 8;
        size_t o = (size_t)mr * GE_D + ncol;
        float val = acc[r] + bs;
        if (act) val = val * (1.f / (1.f + __expf(-val)));
        if (resid) val += resid[o];
        C[o] = val;
    }
    (void)M;
}

// ---------------------------------------------------------------------------
// h init: x @ in_w + in_b + pe(row % N)
__global__ __launch_bounds__(256) void k_input_proj(const float* __restrict__ x,
                                                    const float* __restrict__ inw,
                                                    const float* __restrict__ inb,
                                                    const float* __restrict__ pe,
                                                    float* __restrict__ h) {
    __shared__ float xr[GE_IN];
    int row = blockIdx.x, d = threadIdx.x;
    if (d < GE_IN) xr[d] = x[(size_t)row * GE_IN + d];
    __syncthreads();
    float acc = inb[d];
#pragma unroll
    for (int c = 0; c < GE_IN; ++c) acc += xr[c] * inw[c * GE_D + d];
    h[(size_t)row * GE_D + d] = acc + pe[(size_t)(row & (GE_N - 1)) * GE_D + d];
}

// ---------------------------------------------------------------------------
// LayerNorm (biased var) * g + b, one block per row
__global__ __launch_bounds__(256) void k_layernorm(const float* __restrict__ h,
                                                   const float* __restrict__ g,
                                                   const float* __restrict__ bta,
                                                   float* __restrict__ hn) {
    __shared__ float red[256];
    int row = blockIdx.x, t = threadIdx.x;
    size_t o = (size_t)row * GE_D + t;
    float v = h[o];
    red[t] = v; __syncthreads();
    for (int s = 128; s > 0; s >>= 1) { if (t < s) red[t] += red[t + s]; __syncthreads(); }
    float mean = red[0] * (1.f / GE_D);
    __syncthreads();
    float dv = v - mean;
    red[t] = dv * dv; __syncthreads();
    for (int s = 128; s > 0; s >>= 1) { if (t < s) red[t] += red[t + s]; __syncthreads(); }
    float var = red[0] * (1.f / GE_D);
    hn[o] = dv * rsqrtf(var + 1e-5f) * g[t] + bta[t];
}

// ---------------------------------------------------------------------------
// Sparse neighbor aggregation (exact binary-A einsum) fused with (1+eps)*hn.
// Gathered adjacency slab is copied global->LDS via the async path.
__global__ __launch_bounds__(256) void k_spmm_agg(const float* __restrict__ adj,
                                                  const int* __restrict__ cidx,
                                                  const float* __restrict__ hn,
                                                  const float* __restrict__ beps,
                                                  int blk,
                                                  float* __restrict__ t, int n) {
    __shared__ float arow[256];
    int b = blockIdx.x / n, i = blockIdx.x % n, d = threadIdx.x;
    int gi = cidx[b * GE_N + i];
    const float* adjrow = adj + (size_t)gi * GE_N;
    float acc = 0.f;
    for (int j0 = 0; j0 < n; j0 += 256) {           // n is a multiple of 256
        const float* gp = adjrow + cidx[b * GE_N + j0 + d];
#if GE_ASYNC_B32
        __builtin_amdgcn_global_load_async_to_lds_b32(
            (__attribute__((address_space(1))) int*)gp,
            (__attribute__((address_space(3))) int*)&arow[d], 0, 0);
        __builtin_amdgcn_s_wait_asynccnt(0);
#else
        arow[d] = *gp;
#endif
        __syncthreads();
        for (int q = 0; q < 256; ++q) {             // uniform branch across block
            if (arow[q] != 0.f)
                acc += hn[((size_t)(b * n + j0 + q)) * GE_D + d];
        }
        __syncthreads();
    }
    size_t o = ((size_t)(b * n + i)) * GE_D + d;
    t[o] = (1.f + beps[blk]) * hn[o] + acc;
}

// ---------------------------------------------------------------------------
// score[b,i] = <h[b,i,:], p> / ||p||
__global__ __launch_bounds__(256) void k_score(const float* __restrict__ h,
                                               const float* __restrict__ p,
                                               float* __restrict__ score, int n) {
    __shared__ float r1[256], r2[256];
    int row = blockIdx.x, t = threadIdx.x;
    float pv = p[t], hv = h[(size_t)row * GE_D + t];
    r1[t] = hv * pv; r2[t] = pv * pv; __syncthreads();
    for (int s = 128; s > 0; s >>= 1) {
        if (t < s) { r1[t] += r1[t + s]; r2[t] += r2[t + s]; }
        __syncthreads();
    }
    if (t == 0) score[row] = r1[0] * rsqrtf(r2[0]);
    (void)n;
}

// Top-k (k=n/2) by rank counting (ties by lower index, like lax.top_k),
// ascending selection order, tanh gate, curidx restriction.
__global__ __launch_bounds__(256) void k_pool_select(const float* __restrict__ score,
                                                     const int* __restrict__ cin,
                                                     int* __restrict__ cout,
                                                     int* __restrict__ selsrc,
                                                     float* __restrict__ gate, int n) {
    __shared__ float sc[GE_N];
    __shared__ unsigned char fl[GE_N];
    int b = blockIdx.x, t = threadIdx.x, k = n >> 1;
    for (int i = t; i < n; i += 256) sc[i] = score[b * n + i];
    __syncthreads();
    for (int i = t; i < n; i += 256) {
        float si = sc[i]; int r = 0;
        for (int j = 0; j < n; ++j) {
            float sj = sc[j];
            r += (sj > si) || (sj == si && j < i);
        }
        fl[i] = (r < k) ? 1 : 0;
    }
    __syncthreads();
    for (int i = t; i < n; i += 256) {
        if (fl[i]) {
            int pos = 0;
            for (int j = 0; j < i; ++j) pos += fl[j];
            selsrc[b * GE_N + pos] = i;
            gate[b * GE_N + pos]   = tanhf(sc[i]);
            cout[b * GE_N + pos]   = cin[b * GE_N + i];
        }
    }
}

__global__ __launch_bounds__(256) void k_pool_gather(const float* __restrict__ hin,
                                                     const int* __restrict__ selsrc,
                                                     const float* __restrict__ gate,
                                                     float* __restrict__ hout,
                                                     int n, int k) {
    int b = blockIdx.x / k, pos = blockIdx.x % k, d = threadIdx.x;
    int i = selsrc[b * GE_N + pos];
    hout[((size_t)(b * k + pos)) * GE_D + d] =
        hin[((size_t)(b * n + i)) * GE_D + d] * gate[b * GE_N + pos];
}

// ---------------------------------------------------------------------------
// Readout: per (b,d) mean and unbiased std over 256 nodes -> (B, 512)
__global__ __launch_bounds__(256) void k_final_stats(const float* __restrict__ h,
                                                     float* __restrict__ st) {
    int b = blockIdx.x, d = threadIdx.x;
    float s = 0.f, ss = 0.f;
    for (int i = 0; i < 256; ++i) {
        float v = h[((size_t)(b * 256 + i)) * GE_D + d];
        s += v; ss += v * v;
    }
    float m = s * (1.f / 256.f);
    float var = (ss - 256.f * m * m) * (1.f / 255.f);   // ddof=1
    if (var < 0.f) var = 0.f;
    st[b * 512 + d]       = m;
    st[b * 512 + 256 + d] = sqrtf(var) + 1e-6f;
}

__global__ __launch_bounds__(128) void k_final_linear(const float* __restrict__ st,
                                                      const float* __restrict__ lw,
                                                      const float* __restrict__ lb,
                                                      float* __restrict__ out) {
    int b = blockIdx.x, j = threadIdx.x;
    float a = lb[j];
    for (int c = 0; c < 512; ++c) a += st[b * 512 + c] * lw[c * GE_LAT + j];
    out[b * GE_LAT + j] = a;
}

// ---------------------------------------------------------------------------
extern "C" void kernel_launch(void* const* d_in, const int* in_sizes, int n_in,
                              void* d_out, int out_size, void* d_ws, size_t ws_size,
                              hipStream_t stream) {
    const float* x      = (const float*)d_in[0];
    const float* adj    = (const float*)d_in[1];
    const float* pos    = (const float*)d_in[2];
    const float* in_w   = (const float*)d_in[3];
    const float* in_b   = (const float*)d_in[4];
    const float* pw1    = (const float*)d_in[5];
    const float* pb1    = (const float*)d_in[6];
    const float* pw2    = (const float*)d_in[7];
    const float* pb2    = (const float*)d_in[8];
    const float* bg     = (const float*)d_in[9];
    const float* bb     = (const float*)d_in[10];
    const float* bw1    = (const float*)d_in[11];
    const float* bb1    = (const float*)d_in[12];
    const float* bw2    = (const float*)d_in[13];
    const float* bb2    = (const float*)d_in[14];
    const float* beps   = (const float*)d_in[15];
    const float* pool_p = (const float*)d_in[16];
    const float* lw     = (const float*)d_in[17];
    const float* lb     = (const float*)d_in[18];
    (void)in_sizes; (void)n_in; (void)out_size; (void)ws_size;

    const size_t HSZ = (size_t)GE_B * GE_N * GE_D;   // 2,097,152 floats (8MB)
    float*    bufA   = (float*)d_ws;        // h ping
    float*    bufB   = bufA + HSZ;          // LN out / t1 / pe scratch
    float*    bufC   = bufB + HSZ;          // agg out / u scratch
    float*    bufD   = bufC + HSZ;          // h pong
    float*    score  = bufD + HSZ;          // B*N
    int*      selsrc = (int*)(score + GE_B * GE_N);
    float*    gate   = (float*)(selsrc + GE_B * GE_N);
    int*      cidxA  = (int*)(gate + GE_B * GE_N);
    int*      cidxB  = cidxA + GE_B * GE_N;
    _Float16* wpack  = (_Float16*)(cidxB + GE_B * GE_N);   // 128 KB packed W
    float*    posst  = (float*)(wpack + GE_WPK);           // 8 floats
    float*    fstats = posst + 8;                          // B*512

    // --- prologue: indices, positional embedding, input projection ---------
    k_init_cidx<<<GE_B * GE_N / 256, 256, 0, stream>>>(cidxA);
    k_pos_stats<<<1, 192, 0, stream>>>(pos, posst);
    k_pos_mlp1<<<GE_N, 256, 0, stream>>>(pos, posst, pw1, pb1, bufC);      // u
    {
        dim3 g(GE_N / 16, 4);
        k_pack_w<<<256, 256, 0, stream>>>(pw2, wpack);
        k_gemm256<<<g, 128, 0, stream>>>(bufC, wpack, pb2, nullptr, bufB, // pe
                                         GE_N, 0);
    }
    k_input_proj<<<GE_B * GE_N, 256, 0, stream>>>(x, in_w, in_b, bufB, bufA);

    float* h = bufA; float* hAlt = bufD;
    int* ci = cidxA; int* cn = cidxB;
    int n = GE_N, blk = 0;

    auto run_block = [&](int bidx) {
        int rows = GE_B * n;
        dim3 g(rows / 16, 4);
        k_layernorm<<<rows, 256, 0, stream>>>(h, bg + bidx * GE_D,
                                              bb + bidx * GE_D, bufB);
        k_spmm_agg<<<rows, 256, 0, stream>>>(adj, ci, bufB, beps, bidx, bufC, n);
        k_pack_w<<<256, 256, 0, stream>>>(bw1 + (size_t)bidx * GE_D * GE_D, wpack);
        k_gemm256<<<g, 128, 0, stream>>>(bufC, wpack, bb1 + bidx * GE_D,
                                         nullptr, bufB, rows, 1);
        k_pack_w<<<256, 256, 0, stream>>>(bw2 + (size_t)bidx * GE_D * GE_D, wpack);
        k_gemm256<<<g, 128, 0, stream>>>(bufB, wpack, bb2 + bidx * GE_D,
                                         h, h, rows, 0);
    };

    for (int dep = 0; dep < 3; ++dep) {
        run_block(blk++); run_block(blk++);
        k_score<<<GE_B * n, 256, 0, stream>>>(h, pool_p + dep * GE_D, score, n);
        k_pool_select<<<GE_B, 256, 0, stream>>>(score, ci, cn, selsrc, gate, n);
        int k = n >> 1;
        k_pool_gather<<<GE_B * k, 256, 0, stream>>>(h, selsrc, gate, hAlt, n, k);
        { float* tmp = h; h = hAlt; hAlt = tmp; }
        { int* tmp = ci; ci = cn; cn = tmp; }
        n = k;
    }
    run_block(blk++); run_block(blk++);

    k_final_stats<<<GE_B, 256, 0, stream>>>(h, fstats);
    k_final_linear<<<GE_B, 128, 0, stream>>>(fstats, lw, lb, (float*)d_out);
}
